// SFB_11793980195105
// MI455X (gfx1250) — matrix operand order repaired and checked by reading, not compiled, not run on hardware
//
#include <hip/hip_runtime.h>
#include <hip/hip_bf16.h>

typedef __attribute__((ext_vector_type(16))) _Float16 v16h;
typedef __attribute__((ext_vector_type(8)))  _Float16 v8h;
typedef __attribute__((ext_vector_type(8)))  float    v8f;

#define THITA 0.0001f

// ---------------------------------------------------------------------------
// WMMA helpers (f16 16x16x32, f32 accumulate)
// A fragment: row-major A[m][k] with leading dim ld (halves).
//   lane<16 : row m0+lane,    K = {k0..k0+7, k0+16..k0+23}
//   lane>=16: row m0+lane-16, K = {k0+8..k0+15, k0+24..k0+31}
// B fragment: column-major storage base[(col)*ld + k]  (B is K x N logically)
//   lane<16 : col n0+lane,    K = {k0..k0+7, k0+16..k0+23}
//   lane>=16: col n0+lane-16, K = {k0+8..k0+15, k0+24..k0+31}
// C/D: acc[r] -> D[rbase+r][col],  col=lane&15, rbase=(lane&16)?8:0
// ---------------------------------------------------------------------------
__device__ __forceinline__ v16h load_frag_a(const _Float16* base, int ld, int m0, int k0, int lane) {
    int r  = lane & 15;
    int kk = (lane & 16) ? 8 : 0;
    const _Float16* p = base + (m0 + r) * ld + k0 + kk;
    v8h lo = *(const v8h*)(p);
    v8h hi = *(const v8h*)(p + 16);
    return __builtin_shufflevector(lo, hi, 0,1,2,3,4,5,6,7,8,9,10,11,12,13,14,15);
}

__device__ __forceinline__ v16h load_frag_b(const _Float16* base, int ld, int k0, int n0, int lane) {
    int c  = lane & 15;
    int kk = (lane & 16) ? 8 : 0;
    const _Float16* p = base + (n0 + c) * ld + k0 + kk;
    v8h lo = *(const v8h*)(p);
    v8h hi = *(const v8h*)(p + 16);
    return __builtin_shufflevector(lo, hi, 0,1,2,3,4,5,6,7,8,9,10,11,12,13,14,15);
}

__device__ __forceinline__ v8f wmma_f16(v16h a, v16h b, v8f c) {
    return __builtin_amdgcn_wmma_f32_16x16x32_f16(false, a, false, b, (short)0, c, false, false);
}

// ---------------------------------------------------------------------------
// One conv1x1 layer on a 128-pixel tile:
//   OUT[64][128] = relu(W[64x64] @ IN[64][128] + bias)
// Epilogue destinations selected at compile time (no per-element branches).
// ---------------------------------------------------------------------------
template<bool LDSOUT, bool G32OUT, bool G16OUT>
__device__ __forceinline__ void conv_gemm64x128(
    const _Float16* sW, const _Float16* sIn, const float* sBias,
    _Float16* sOut,
    float* g32, long g32rs, long g32cs,
    _Float16* g16, long g16rs, long g16cs,
    int tid)
{
    int wave = tid >> 5, lane = tid & 31;
    int col = lane & 15;
    int rbase = (lane & 16) ? 8 : 0;
#pragma unroll
    for (int i = 0; i < 4; ++i) {
        int t = wave * 4 + i;
        int mt = t >> 3, nt = t & 7;
        v8f acc = {};
        v16h a0 = load_frag_a(sW, 64, mt * 16, 0, lane);
        v16h b0 = load_frag_b(sIn, 64, 0, nt * 16, lane);
        v16h a1 = load_frag_a(sW, 64, mt * 16, 32, lane);
        v16h b1 = load_frag_b(sIn, 64, 32, nt * 16, lane);
        acc = wmma_f16(a0, b0, acc);
        acc = wmma_f16(a1, b1, acc);
#pragma unroll
        for (int r = 0; r < 8; ++r) {
            int m = mt * 16 + rbase + r;   // output channel
            int n = nt * 16 + col;         // pixel within tile
            float v = acc[r] + sBias[m];
            v = fmaxf(v, 0.f);
            if (LDSOUT) sOut[n * 64 + m] = (_Float16)v;
            if (G32OUT) g32[(long)m * g32rs + (long)n * g32cs] = v;
            if (G16OUT) g16[(long)m * g16rs + (long)n * g16cs] = (_Float16)v;
        }
    }
}

// ---------------------------------------------------------------------------
// Kernel 1: x chain.  x -> x1 (f32 out + f16 LDS) -> x2 -> q (qt[n][c]),
//                                                         v (vcm[c][n])
// grid (18, 4): 18 tiles of 128 pixels, 4 batches. block 256 (8 waves).
// ---------------------------------------------------------------------------
__global__ __launch_bounds__(256) void conv_x_kernel(
    const float* __restrict__ x,
    const float* __restrict__ Wx0, const float* __restrict__ bx0,
    const float* __restrict__ Wx1, const float* __restrict__ bx1,
    const float* __restrict__ Wq,  const float* __restrict__ bq,
    const float* __restrict__ Wv,  const float* __restrict__ bv,
    float* __restrict__ x1g, _Float16* __restrict__ qt, _Float16* __restrict__ vcm)
{
    __shared__ alignas(32) _Float16 sW[64 * 64];
    __shared__ float sBias[64];
    __shared__ alignas(32) _Float16 sIn[128 * 64];
    __shared__ alignas(32) _Float16 sT1[128 * 64];
    __shared__ alignas(32) _Float16 sT2[128 * 64];

    const long N = 2304;
    int tid = threadIdx.x;
    int b = blockIdx.y;
    int p0 = blockIdx.x * 128;

    for (int idx = tid; idx < 8192; idx += 256) {
        int c = idx >> 7, p = idx & 127;
        sIn[p * 64 + c] = (_Float16)x[((long)b * 64 + c) * N + p0 + p];
    }
    for (int idx = tid; idx < 4096; idx += 256) sW[idx] = (_Float16)Wx0[idx];
    if (tid < 64) sBias[tid] = bx0[tid];
    __syncthreads();
    conv_gemm64x128<true, true, false>(sW, sIn, sBias, sT1,
                    x1g + (long)b * 64 * N + p0, N, 1, nullptr, 0, 0, tid);
    __syncthreads();
    for (int idx = tid; idx < 4096; idx += 256) sW[idx] = (_Float16)Wx1[idx];
    if (tid < 64) sBias[tid] = bx1[tid];
    __syncthreads();
    conv_gemm64x128<true, false, false>(sW, sT1, sBias, sT2,
                    nullptr, 0, 0, nullptr, 0, 0, tid);
    __syncthreads();
    for (int idx = tid; idx < 4096; idx += 256) sW[idx] = (_Float16)Wq[idx];
    if (tid < 64) sBias[tid] = bq[tid];
    __syncthreads();
    conv_gemm64x128<false, false, true>(sW, sT2, sBias, nullptr, nullptr, 0, 0,
                    qt + ((long)b * N + p0) * 64, 1, 64, tid);
    __syncthreads();
    for (int idx = tid; idx < 4096; idx += 256) sW[idx] = (_Float16)Wv[idx];
    if (tid < 64) sBias[tid] = bv[tid];
    __syncthreads();
    conv_gemm64x128<false, false, true>(sW, sT2, sBias, nullptr, nullptr, 0, 0,
                    vcm + (long)b * 64 * N + p0, N, 1, tid);
}

// ---------------------------------------------------------------------------
// Kernel 2: event chain with pre-subsampling (::16,::16).
// event(B,64,768,768) -> ef0 -> ef1 -> k stored transposed kt[b][m][c]
// ---------------------------------------------------------------------------
__global__ __launch_bounds__(256) void conv_e_kernel(
    const float* __restrict__ event,
    const float* __restrict__ We0, const float* __restrict__ be0,
    const float* __restrict__ We1, const float* __restrict__ be1,
    const float* __restrict__ Wk,  const float* __restrict__ bk,
    _Float16* __restrict__ kt)
{
    __shared__ alignas(32) _Float16 sW[64 * 64];
    __shared__ float sBias[64];
    __shared__ alignas(32) _Float16 sIn[128 * 64];
    __shared__ alignas(32) _Float16 sT1[128 * 64];
    __shared__ alignas(32) _Float16 sT2[128 * 64];

    const long N = 2304;
    int tid = threadIdx.x;
    int b = blockIdx.y;
    int p0 = blockIdx.x * 128;

    for (int idx = tid; idx < 8192; idx += 256) {
        int c = idx >> 7, p = idx & 127;
        int pg = p0 + p;
        int i = pg / 48, j = pg - i * 48;
        long e = ((long)(b * 64 + c) * 768 + (long)i * 16) * 768 + (long)j * 16;
        sIn[p * 64 + c] = (_Float16)event[e];
    }
    for (int idx = tid; idx < 4096; idx += 256) sW[idx] = (_Float16)We0[idx];
    if (tid < 64) sBias[tid] = be0[tid];
    __syncthreads();
    conv_gemm64x128<true, false, false>(sW, sIn, sBias, sT1,
                    nullptr, 0, 0, nullptr, 0, 0, tid);
    __syncthreads();
    for (int idx = tid; idx < 4096; idx += 256) sW[idx] = (_Float16)We1[idx];
    if (tid < 64) sBias[tid] = be1[tid];
    __syncthreads();
    conv_gemm64x128<true, false, false>(sW, sT1, sBias, sT2,
                    nullptr, 0, 0, nullptr, 0, 0, tid);
    __syncthreads();
    for (int idx = tid; idx < 4096; idx += 256) sW[idx] = (_Float16)Wk[idx];
    if (tid < 64) sBias[tid] = bk[tid];
    __syncthreads();
    conv_gemm64x128<false, false, true>(sW, sT2, sBias, nullptr, nullptr, 0, 0,
                    kt + ((long)b * N + p0) * 64, 1, 64, tid);
}

// ---------------------------------------------------------------------------
// Kernel 3: energy[b][n][m] = sum_c qt[n][c] * kt[m][c]   (f16 out)
// Both stagings are fully contiguous uint4 (B128) copies.
// grid (18, 36, 4): 128-wide m tile, 64-row n tile, batch
// ---------------------------------------------------------------------------
__global__ __launch_bounds__(256) void energy_kernel(
    const _Float16* __restrict__ qt, const _Float16* __restrict__ kt,
    _Float16* __restrict__ en)
{
    __shared__ alignas(32) _Float16 sQ[64 * 64];   // [n_local][c]  (A)
    __shared__ alignas(32) _Float16 sK[128 * 64];  // [m_local][c]  (B col-major)

    const long N = 2304;
    int tid = threadIdx.x;
    int m0 = blockIdx.x * 128;
    int n0 = blockIdx.y * 64;
    int b  = blockIdx.z;

    const uint4* qb = (const uint4*)(qt + ((long)b * N + n0) * 64);
#pragma unroll
    for (int j = 0; j < 2; ++j) ((uint4*)sQ)[tid * 2 + j] = qb[tid * 2 + j];
    const uint4* kb = (const uint4*)(kt + ((long)b * N + m0) * 64);
#pragma unroll
    for (int j = 0; j < 4; ++j) ((uint4*)sK)[tid * 4 + j] = kb[tid * 4 + j];
    __syncthreads();

    int wave = tid >> 5, lane = tid & 31;
    int col = lane & 15, rbase = (lane & 16) ? 8 : 0;
    _Float16* eb = en + ((long)b * N + n0) * N + m0;
#pragma unroll
    for (int i = 0; i < 4; ++i) {
        int t = wave * 4 + i;
        int ntile = t >> 3, mtile = t & 7;
        v8f acc = {};
        v16h a0 = load_frag_a(sQ, 64, ntile * 16, 0, lane);
        v16h b0 = load_frag_b(sK, 64, 0, mtile * 16, lane);
        v16h a1 = load_frag_a(sQ, 64, ntile * 16, 32, lane);
        v16h b1 = load_frag_b(sK, 64, 32, mtile * 16, lane);
        acc = wmma_f16(a0, b0, acc);
        acc = wmma_f16(a1, b1, acc);
#pragma unroll
        for (int r = 0; r < 8; ++r) {
            int n = ntile * 16 + rbase + r;
            int m = mtile * 16 + col;
            eb[(long)n * N + m] = (_Float16)acc[r];
        }
    }
}

// ---------------------------------------------------------------------------
// Kernel 4: row softmax in place (f16). One block per (b, n) row, 256 thr.
// exp(x - max) <= 1, so unnormalized f16 storage is safe. attn is L2-resident.
// ---------------------------------------------------------------------------
__global__ __launch_bounds__(256) void softmax_kernel(_Float16* __restrict__ attn)
{
    __shared__ float sred[256];
    const long N = 2304;
    int tid = threadIdx.x;
    _Float16* p = attn + ((long)blockIdx.y * N + blockIdx.x) * N;

    float mx = -3.0e38f;
    for (int i = tid; i < N; i += 256) mx = fmaxf(mx, (float)p[i]);
    sred[tid] = mx; __syncthreads();
    for (int s = 128; s > 0; s >>= 1) {
        if (tid < s) sred[tid] = fmaxf(sred[tid], sred[tid + s]);
        __syncthreads();
    }
    mx = sred[0]; __syncthreads();

    float sum = 0.f;
    for (int i = tid; i < N; i += 256) {
        float e = __expf((float)p[i] - mx);
        p[i] = (_Float16)e;
        sum += e;
    }
    sred[tid] = sum; __syncthreads();
    for (int s = 128; s > 0; s >>= 1) {
        if (tid < s) sred[tid] += sred[tid + s];
        __syncthreads();
    }
    float inv = 1.f / sred[0];
    for (int i = tid; i < N; i += 256) p[i] = (_Float16)((float)p[i] * inv);
}

// ---------------------------------------------------------------------------
// Kernel 5: D[m][c] = sum_n attn[m][n] * vcm[c][n];
//           out[b][c][m] = THITA*gamma*D[m][c] + x1[b][c][m]
// Double-buffered LDS: global loads for chunk kb+1 issued before the WMMAs
// on chunk kb, ds_store after, so VMEM overlaps the matrix pipe.
// grid (18, 4): 128 m-rows per block; K loop over 36 chunks of 64 n.
// ---------------------------------------------------------------------------
__global__ __launch_bounds__(256) void attnout_kernel(
    const _Float16* __restrict__ attn, const _Float16* __restrict__ vcm,
    const float* __restrict__ x1g, const float* __restrict__ gamma,
    float* __restrict__ out)
{
    __shared__ alignas(32) _Float16 sA[2][128 * 64];  // attn chunk [m_local][n_local]
    __shared__ alignas(32) _Float16 sB[2][64 * 64];   // v chunk  [c][n_local]

    const long N = 2304;
    int tid = threadIdx.x;
    int m0 = blockIdx.x * 128;
    int b  = blockIdx.y;
    int wave = tid >> 5, lane = tid & 31;

    const _Float16* ab_base = attn + ((long)b * N + m0) * N;  // [m_local][n_glob]
    const _Float16* vb_base = vcm + (long)b * 64 * N;         // [c][n_glob]

    // staging geometry: 32 halves/thread of A, 16 halves/thread of B per chunk
    int aml = (tid * 32) >> 6, anl = (tid * 32) & 63;
    int bc  = (tid * 16) >> 6, bnl = (tid * 16) & 63;

    uint4 ra[4], rb[2];
    // preload chunk 0
    {
        const uint4* sa = (const uint4*)(ab_base + (long)aml * N + anl);
#pragma unroll
        for (int j = 0; j < 4; ++j) ra[j] = sa[j];
        const uint4* sb = (const uint4*)(vb_base + (long)bc * N + bnl);
#pragma unroll
        for (int j = 0; j < 2; ++j) rb[j] = sb[j];
        uint4* da = (uint4*)(&sA[0][tid * 32]);
#pragma unroll
        for (int j = 0; j < 4; ++j) da[j] = ra[j];
        uint4* db = (uint4*)(&sB[0][tid * 16]);
#pragma unroll
        for (int j = 0; j < 2; ++j) db[j] = rb[j];
    }
    __syncthreads();

    v8f acc[4];
#pragma unroll
    for (int i = 0; i < 4; ++i) acc[i] = (v8f){};

    for (int kb = 0; kb < 36; ++kb) {
        int cur = kb & 1;
        bool more = (kb + 1) < 36;
        if (more) {
            int nb = (kb + 1) * 64;
            const uint4* sa = (const uint4*)(ab_base + (long)aml * N + nb + anl);
#pragma unroll
            for (int j = 0; j < 4; ++j) ra[j] = sa[j];
            const uint4* sb = (const uint4*)(vb_base + (long)bc * N + nb + bnl);
#pragma unroll
            for (int j = 0; j < 2; ++j) rb[j] = sb[j];
        }
#pragma unroll
        for (int i = 0; i < 4; ++i) {
            int t = wave * 4 + i;
            int mtile = t >> 2, ctile = t & 3;   // 8 m-tiles x 4 c-tiles
            v16h a0 = load_frag_a(sA[cur], 64, mtile * 16, 0, lane);
            v16h b0 = load_frag_b(sB[cur], 64, 0, ctile * 16, lane);
            v16h a1 = load_frag_a(sA[cur], 64, mtile * 16, 32, lane);
            v16h b1 = load_frag_b(sB[cur], 64, 32, ctile * 16, lane);
            acc[i] = wmma_f16(a0, b0, acc[i]);
            acc[i] = wmma_f16(a1, b1, acc[i]);
        }
        if (more) {
            uint4* da = (uint4*)(&sA[cur ^ 1][tid * 32]);
#pragma unroll
            for (int j = 0; j < 4; ++j) da[j] = ra[j];
            uint4* db = (uint4*)(&sB[cur ^ 1][tid * 16]);
#pragma unroll
            for (int j = 0; j < 2; ++j) db[j] = rb[j];
        }
        __syncthreads();
    }

    float scale = THITA * gamma[0];
    int col = lane & 15, rbase = (lane & 16) ? 8 : 0;
#pragma unroll
    for (int i = 0; i < 4; ++i) {
        int t = wave * 4 + i;
        int mtile = t >> 2, ctile = t & 3;
#pragma unroll
        for (int r = 0; r < 8; ++r) {
            int m = m0 + mtile * 16 + rbase + r;
            int c = ctile * 16 + col;
            long oidx = ((long)b * 64 + c) * N + m;
            out[oidx] = scale * acc[i][r] + x1g[oidx];
        }
    }
}

// ---------------------------------------------------------------------------
extern "C" void kernel_launch(void* const* d_in, const int* in_sizes, int n_in,
                              void* d_out, int out_size, void* d_ws, size_t ws_size,
                              hipStream_t stream) {
    const float* x     = (const float*)d_in[0];
    const float* event = (const float*)d_in[1];
    const float* Wx0 = (const float*)d_in[2],  *bx0 = (const float*)d_in[3];
    const float* Wx1 = (const float*)d_in[4],  *bx1 = (const float*)d_in[5];
    const float* We0 = (const float*)d_in[6],  *be0 = (const float*)d_in[7];
    const float* We1 = (const float*)d_in[8],  *be1 = (const float*)d_in[9];
    const float* Wq  = (const float*)d_in[10], *bq  = (const float*)d_in[11];
    const float* Wk  = (const float*)d_in[12], *bk  = (const float*)d_in[13];
    const float* Wv  = (const float*)d_in[14], *bv  = (const float*)d_in[15];
    const float* gamma = (const float*)d_in[16];
    float* out = (float*)d_out;

    const long N = 2304, B = 4, C = 64;
    char* ws = (char*)d_ws;
    float*    x1g  = (float*)ws;     ws += B * C * N * sizeof(float);   // 2.36 MB
    _Float16* qt   = (_Float16*)ws;  ws += B * N * C * 2;               // 1.18 MB
    _Float16* kt   = (_Float16*)ws;  ws += B * N * C * 2;               // 1.18 MB
    _Float16* vcm  = (_Float16*)ws;  ws += B * C * N * 2;               // 1.18 MB
    _Float16* attn = (_Float16*)ws;                                     // 42.5 MB

    dim3 blk(256);
    conv_x_kernel<<<dim3(18, 4), blk, 0, stream>>>(x, Wx0, bx0, Wx1, bx1,
                                                   Wq, bq, Wv, bv, x1g, qt, vcm);
    conv_e_kernel<<<dim3(18, 4), blk, 0, stream>>>(event, We0, be0, We1, be1,
                                                   Wk, bk, kt);
    energy_kernel<<<dim3(18, 36, 4), blk, 0, stream>>>(qt, kt, attn);
    softmax_kernel<<<dim3(2304, 4), blk, 0, stream>>>(attn);
    attnout_kernel<<<dim3(18, 4), blk, 0, stream>>>(attn, vcm, x1g, gamma, out);
}
